// CRF_38826504356651
// MI455X (gfx1250) — matrix operand order repaired and checked by reading, not compiled, not run on hardware
//
#include <hip/hip_runtime.h>
#include <cstdint>
#include <cstddef>

#define FILLV (-1000.0f)

typedef _Float16 v16h __attribute__((ext_vector_type(16)));
typedef float    v8f  __attribute__((ext_vector_type(8)));
typedef unsigned int v4u  __attribute__((ext_vector_type(4)));
typedef int      v8i  __attribute__((ext_vector_type(8)));
typedef int      v4i  __attribute__((ext_vector_type(4)));

namespace {
constexpr int kB = 512;      // batch
constexpr int kT = 512;      // time steps
constexpr int kL = 126;      // labels
constexpr int kS = 128;      // states = L+2  (fits WMMA K/N tiling exactly)
constexpr int kRows = 16;    // batch rows per workgroup (WMMA M)
constexpr int kThreads = 256; // 8 waves of 32
// LDS layout (floats/dwords):
//   Tt      [128][128] f32 transposed transitions  (64 KB)
//   pre_s   [16][128]  f32 normalized forward state (8 KB)
//   Mlds    [16][128]  f32 max-plus result          (8 KB)
//   tile0/1 [16][126]  f32 TDM double buffers       (~16 KB)
//   ETb     8192 dw    f16x2 exp(T) WMMA-B frags    (32 KB)
//   Af      1024 dw    f16x2 exp(pre) WMMA-A frags  (4 KB)
//   Slds/LS/accv 3*16 f32
constexpr int kSmemBytes =
    (kS * kS + 2 * kRows * kS + 2 * kRows * kL) * 4 + 8192 * 4 + 1024 * 4 + 3 * 16 * 4;
}

// ---- Tensor Data Mover: 2-D tile load (16 rows x 126 f32, row stride T*L) ----
__device__ __forceinline__ void tdm_load_tile(const float* gsrc, unsigned lds_off) {
  unsigned long long ga = (unsigned long long)(uintptr_t)gsrc;
  // D# group0: count=1, lds_addr, global_addr[56:0], type=2
  v4u g0 = { 1u, lds_off, (unsigned)ga, (unsigned)(ga >> 32) | (2u << 30) };
  // D# group1: data_size=4B, tensor_dim0=126, tensor_dim1=16, tile_dim0=126,
  //            tile_dim1=16, tensor_dim0_stride = T*L = 64512 elements
  v8i g1 = { (int)0x00020000,
             (int)(126u << 16),
             (int)(16u  << 16),
             (int)(126u << 16),
             16,
             (int)(kT * kL),
             0, 0 };
  v4i g2 = { 0, 0, 0, 0 };
  v4i g3 = { 0, 0, 0, 0 };
  v8i g4 = { 0, 0, 0, 0, 0, 0, 0, 0 };  // clang-23 toolchain: extra 8-dword group
  __builtin_amdgcn_tensor_load_to_lds(g0, g1, g2, g3, g4, 0);
}

__global__ __launch_bounds__(kThreads)
void crf_fwd_kernel(const float* __restrict__ emissions,
                    const int*   __restrict__ labels,
                    const float* __restrict__ trans,
                    float* __restrict__ out) {
  extern __shared__ __align__(16) char smem_raw[];
  float*    Tt    = (float*)smem_raw;            // Tt[j*128+i] = T[i][j]
  float*    pre_s = Tt + kS * kS;                // [16][128]
  float*    Mlds  = pre_s + kRows * kS;          // [16][128]
  float*    tile0 = Mlds + kRows * kS;           // [16][126]
  float*    tile1 = tile0 + kRows * kL;          // [16][126]
  unsigned* ETb   = (unsigned*)(tile1 + kRows * kL); // 8192 dw
  unsigned* Af    = ETb + 8192;                  // 1024 dw
  float*    Slds  = (float*)(Af + 1024);         // [16]
  float*    LS    = Slds + 16;                   // [16]
  float*    accv  = LS + 16;                     // [16]

  const int tid  = threadIdx.x;
  const int lane = tid & 31;
  const int wv   = tid >> 5;              // wave id 0..7
  const int jmp  = (wv << 4) + (lane & 15); // this lane's output column
  const int bh   = lane >> 4;             // batch half within wave
  const int b0   = blockIdx.x * kRows;

  // ---------------- init: build Tt (transposed) and ETb = f16 exp(T) B-frags ----
  for (int o = tid; o < kS * kS; o += kThreads) {
    int j = o >> 7, i = o & 127;
    Tt[o] = trans[i * kS + j];
  }
  // B fragment layout (16-bit, 32x16): lanes 0-15 K=0..7,16..23 ; lanes 16-31 K=8..15,24..31
  for (int idx = tid; idx < 8192; idx += kThreads) {
    int v = idx & 7, ln = (idx >> 3) & 31, kc = (idx >> 8) & 3, w = idx >> 10;
    int n  = (w << 4) + (ln & 15);
    int k0 = kc * 32 + (v < 4 ? 2 * v : 16 + 2 * (v - 4)) + ((ln >> 4) << 3);
    float e0 = __expf(trans[k0 * kS + n]);
    float e1 = __expf(trans[(k0 + 1) * kS + n]);
    union { unsigned u; _Float16 h[2]; } pk;
    pk.h[0] = (_Float16)e0; pk.h[1] = (_Float16)e1;
    ETb[idx] = pk.u;
  }
  // initial pre = E[0] (START row), already max-normalized (max = 0 at state 126)
  for (int idx = tid; idx < kRows * kS; idx += kThreads)
    pre_s[idx] = ((idx & 127) == kL) ? 0.0f : FILLV;
  if (tid < 16) accv[tid] = 0.0f;

  // kick off first emissions tile (t=1 -> buffer 1)
  if (wv == 0) {
    tdm_load_tile(emissions + ((size_t)b0 * kT + 0) * kL,
                  (unsigned)(uintptr_t)tile1);
  }
  __syncthreads();

  // ---------------- the scan: t = 1 .. T+1 ------------------------------------
  for (int t = 1; t <= kT + 1; ++t) {
    // prefetch next tile via TDM; wait so tile[t&1] is resident
    if (wv == 0) {
      if (t + 1 <= kT) {
        tdm_load_tile(emissions + ((size_t)b0 * kT + (size_t)t) * kL,
                      (unsigned)(uintptr_t)(((t + 1) & 1) ? tile1 : tile0));
        __builtin_amdgcn_s_wait_tensorcnt(1);
      } else {
        __builtin_amdgcn_s_wait_tensorcnt(0);
      }
    }
    // build A fragments: Af = f16 exp(pre_s) in WMMA A layout (K-chunks of 32)
    #pragma unroll
    for (int u = 0; u < 4; ++u) {
      int idx = tid * 4 + u;
      int v = idx & 7, ln = (idx >> 3) & 31, kc = idx >> 8;
      int b  = ln & 15;
      int k0 = kc * 32 + (v < 4 ? 2 * v : 16 + 2 * (v - 4)) + ((ln >> 4) << 3);
      float e0 = __expf(pre_s[b * kS + k0]);
      float e1 = __expf(pre_s[b * kS + k0 + 1]);
      union { unsigned uu; _Float16 h[2]; } pk;
      pk.h[0] = (_Float16)e0; pk.h[1] = (_Float16)e1;
      Af[idx] = pk.uu;
    }
    if (tid < 16) Slds[tid] = 0.0f;
    __syncthreads();

    // --- max-plus: M[b][j] = max_i (pre[b][i] + T[i][j]); lane owns column jmp,
    //     8 batch rows (bh*8 .. bh*8+7), float4 LDS traffic, no global traffic.
    {
      float mbb[8];
      #pragma unroll
      for (int bb = 0; bb < 8; ++bb) mbb[bb] = -3.0e38f;
      const float4* trow = (const float4*)(Tt + jmp * kS);
      const float4* prow[8];
      #pragma unroll
      for (int bb = 0; bb < 8; ++bb)
        prow[bb] = (const float4*)(pre_s + (bh * 8 + bb) * kS);
      #pragma unroll 4
      for (int i4 = 0; i4 < kS / 4; ++i4) {
        float4 tv = trow[i4];
        #pragma unroll
        for (int bb = 0; bb < 8; ++bb) {
          float4 pv = prow[bb][i4];
          float c0 = fmaxf(pv.x + tv.x, pv.y + tv.y);
          float c1 = fmaxf(pv.z + tv.z, pv.w + tv.w);
          mbb[bb] = fmaxf(mbb[bb], fmaxf(c0, c1));
        }
      }
      #pragma unroll
      for (int bb = 0; bb < 8; ++bb) Mlds[(bh * 8 + bb) * kS + jmp] = mbb[bb];
    }

    // --- WMMA: Q(16x16 tile, cols 16w..) = exp(pre)[16x128] x exp(T)[128x16]
    v8f cacc = {};
    #pragma unroll
    for (int kc = 0; kc < 4; ++kc) {
      v4u a0 = *(const v4u*)(Af + (kc * 32 + lane) * 8);
      v4u a1 = *(const v4u*)(Af + (kc * 32 + lane) * 8 + 4);
      v4u bq0 = *(const v4u*)(ETb + ((wv * 4 + kc) * 32 + lane) * 8);
      v4u bq1 = *(const v4u*)(ETb + ((wv * 4 + kc) * 32 + lane) * 8 + 4);
      union { v4u u[2]; v16h h; } ua, ub;
      ua.u[0] = a0; ua.u[1] = a1; ub.u[0] = bq0; ub.u[1] = bq1;
      cacc = __builtin_amdgcn_wmma_f32_16x16x32_f16(
          false, ua.h, false, ub.h, (short)0, cacc, false, false);
    }

    // --- S[b] += sum_j q[b][j] * exp(-M[b][j]); same-wave Mlds (in-order LDS)
    #pragma unroll
    for (int r = 0; r < 8; ++r) {
      int b = r + (bh << 3);
      float q  = cacc[r];
      float Mv = Mlds[b * kS + jmp];
      float contrib = (q > 0.0f) ? __expf(__logf(q) - Mv) : 0.0f;
      #pragma unroll
      for (int off = 8; off >= 1; off >>= 1)
        contrib += __shfl_xor(contrib, off, 16);
      if ((lane & 15) == 0) atomicAdd(&Slds[b], contrib);
    }
    __syncthreads();
    if (tid < 16) LS[tid] = __logf(Slds[tid]);
    __syncthreads();

    // --- pre_new = obs + M + logS ; renormalize by row max, fold into accv
    {
      const int bq = tid >> 4;                 // 16 threads per batch row
      const float* tl = (t & 1) ? tile1 : tile0;
      const float ls = LS[bq];
      float pn[8];
      float pmax = -3.0e38f;
      #pragma unroll
      for (int u = 0; u < 8; ++u) {
        int idx = tid * 8 + u;
        int j = idx & 127;
        float obs;
        if (t <= kT) obs = (j < kL) ? tl[bq * kL + j] : FILLV;   // E[1..T]
        else         obs = (j == kL + 1) ? 0.0f : FILLV;         // END row
        float v = obs + Mlds[idx] + ls;
        pn[u] = v;
        pmax = fmaxf(pmax, v);
      }
      #pragma unroll
      for (int off = 8; off >= 1; off >>= 1)
        pmax = fmaxf(pmax, __shfl_xor(pmax, off, 16));
      #pragma unroll
      for (int u = 0; u < 8; ++u) pre_s[tid * 8 + u] = pn[u] - pmax;
      if ((tid & 15) == 0) accv[bq] += pmax;
    }
    __syncthreads();
  }

  // ---------------- total path score per row: acc + logsumexp(pre_s) ----------
  const int bq = tid >> 4;
  const int sl = tid & 15;
  float es = 0.0f;
  #pragma unroll
  for (int u = 0; u < 8; ++u) es += __expf(pre_s[tid * 8 + u]);
  #pragma unroll
  for (int off = 8; off >= 1; off >>= 1) es += __shfl_xor(es, off, 16);
  float total_b = accv[bq] + __logf(es);   // valid on all 16 lanes of the group

  // ---------------- real (labeled) path score ---------------------------------
  {
    const int gb = b0 + bq;
    const float* Eb = emissions + (size_t)gb * kT * kL;
    const int*   Lb = labels + (size_t)gb * kT;
    float real = 0.0f;
    for (int t = sl; t < kT; t += 16) {
      int lab = Lb[t];
      real += Eb[t * kL + lab];
      if (t > 0) { int lp = Lb[t - 1]; real += Tt[lab * kS + lp]; } // T[lp][lab]
    }
    if (sl == 0) {
      int l0 = Lb[0];       real += Tt[l0 * kS + kL];           // T[START][l0]
      int le = Lb[kT - 1];  real += Tt[(kL + 1) * kS + le];     // T[le][END]
    }
    #pragma unroll
    for (int off = 8; off >= 1; off >>= 1) real += __shfl_xor(real, off, 16);
    if (sl == 0) atomicAdd(out, total_b - real);
  }
}

extern "C" void kernel_launch(void* const* d_in, const int* in_sizes, int n_in,
                              void* d_out, int out_size, void* d_ws, size_t ws_size,
                              hipStream_t stream) {
  const float* emissions = (const float*)d_in[0];
  const int*   labels    = (const int*)d_in[1];
  const float* trans     = (const float*)d_in[2];
  float* out = (float*)d_out;
  (void)in_sizes; (void)n_in; (void)out_size; (void)d_ws; (void)ws_size;

  (void)hipMemsetAsync(out, 0, sizeof(float), stream);
  (void)hipFuncSetAttribute(reinterpret_cast<const void*>(crf_fwd_kernel),
                            hipFuncAttributeMaxDynamicSharedMemorySize, kSmemBytes);
  crf_fwd_kernel<<<kB / kRows, kThreads, kSmemBytes, stream>>>(
      emissions, labels, trans, out);
}
